// MultiHeadSelfAttention_4372276707931
// MI455X (gfx1250) — matrix operand order repaired and checked by reading, not compiled, not run on hardware
//
#include <hip/hip_runtime.h>
#include <hip/hip_bf16.h>
#include <math.h>

// ---------------------------------------------------------------------------
// MHA forward for MI455X (gfx1250, wave32, WMMA).
// Stage 1: cast f32 -> f16
// Stage 2: fused QKV GEMM (v_wmma_f32_16x16x32_f16, 32x64 wave tile)
//          + degenerate-RoPE epilogue
// Stage 3: flash attention, WMMA for QK^T and PV, LDS round-trip for P,
//          global_prefetch for next K/V tiles.
// ---------------------------------------------------------------------------

typedef __attribute__((ext_vector_type(16))) _Float16 v16h;
typedef __attribute__((ext_vector_type(8)))  _Float16 v8h;
typedef __attribute__((ext_vector_type(4)))  _Float16 v4h;
typedef __attribute__((ext_vector_type(8)))  float    v8f;

#define D_MODEL 1024
#define N_HEADS 16
#define HEAD_DIM 64
#define BATCH 4
#define SEQ 2048
#define ROWS (BATCH * SEQ)   // 8192

// ---- WMMA fragment loaders (layouts per CDNA5 ISA 7.12.2) ------------------
// A-matrix 16x32 f16: lane (m, hi): elem e<8 -> K = 8*hi+e ; e>=8 -> K = 16+8*hi+(e-8)
__device__ inline v16h load_a_frag(const _Float16* __restrict__ base, int ld,
                                   int row, int hi, int k0) {
  const _Float16* p = base + (size_t)row * ld + k0 + 8 * hi;
  v8h lo = *(const v8h*)(p);
  v8h hv = *(const v8h*)(p + 16);
  v16h r;
#pragma unroll
  for (int i = 0; i < 8; ++i) { r[i] = lo[i]; r[8 + i] = hv[i]; }
  return r;
}

// B-matrix 32x16 f16: lane (n, hi): elem e<8 -> K = 8*hi+e ; e>=8 -> K = 16+8*hi+(e-8)  (column n = lane&15)
// "colrow" is the row index in the stored (row-major) matrix holding B's column n.
__device__ inline v16h load_b_frag(const _Float16* __restrict__ base, int ld,
                                   int colrow, int hi, int k0) {
  const _Float16* p = base + (size_t)colrow * ld + k0 + 8 * hi;
  v8h lo = *(const v8h*)(p);
  v8h hv = *(const v8h*)(p + 16);
  v16h r;
#pragma unroll
  for (int i = 0; i < 8; ++i) { r[i] = lo[i]; r[8 + i] = hv[i]; }
  return r;
}

__device__ inline v8f wmma_f16(v16h a, v16h b, v8f c) {
  return __builtin_amdgcn_wmma_f32_16x16x32_f16(false, a, false, b, (short)0, c,
                                                false, false);
}

// ---- Stage 1: cast -------------------------------------------------------
__global__ void cast_f32_f16(const float* __restrict__ in,
                             _Float16* __restrict__ out, int n4) {
  int i = blockIdx.x * blockDim.x + threadIdx.x;
  if (i < n4) {
    float4 v = ((const float4*)in)[i];
    v4h o;
    o[0] = (_Float16)v.x; o[1] = (_Float16)v.y;
    o[2] = (_Float16)v.z; o[3] = (_Float16)v.w;
    ((v4h*)out)[i] = o;
  }
}

// ---- Stage 2: QKV projection + RoPE --------------------------------------
// grid: (ROWS/128, D_MODEL/64, 3{q,k,v}), block: 128 (4 waves).
// Wave computes a 32x64 output tile (two 16-row sub-tiles sharing B frags);
// K loop over 1024 in steps of 32 -> 8 WMMA per k-step.
__global__ void qkv_gemm_rope(const _Float16* __restrict__ xh,
                              const _Float16* __restrict__ Wh, // 3x[1024][1024]
                              _Float16* __restrict__ Qh,       // [B,H,S,Dh]
                              _Float16* __restrict__ Kh,       // [B,H,S,Dh]
                              _Float16* __restrict__ Vt) {     // [B,H,Dh,S]
  const int lane = threadIdx.x & 31;
  const int wave = threadIdx.x >> 5;
  const int m16 = lane & 15;
  const int hi = lane >> 4;
  const int row0 = blockIdx.x * 128 + wave * 32;
  const int n0 = blockIdx.y * 64;
  const int z = blockIdx.z;
  const _Float16* W = Wh + (size_t)z * D_MODEL * D_MODEL;

  v8f acc[2][4] = {};
#pragma unroll 2
  for (int k0 = 0; k0 < D_MODEL; k0 += 32) {
    v16h a0 = load_a_frag(xh, D_MODEL, row0 + m16, hi, k0);
    v16h a1 = load_a_frag(xh, D_MODEL, row0 + 16 + m16, hi, k0);
#pragma unroll
    for (int f = 0; f < 4; ++f) {
      v16h b = load_b_frag(W, D_MODEL, n0 + f * 16 + m16, hi, k0);
      acc[0][f] = wmma_f16(a0, b, acc[0][f]);
      acc[1][f] = wmma_f16(a1, b, acc[1][f]);
    }
  }

  // Epilogue: degenerate RoPE scale (depends only on batch b and channel d):
  //   out[d] = x[d]*(cos(a)+sin(a)) for d<32, x[d]*(cos(a)-sin(a)) for d>=32,
  //   a = b * 10000^(-(d%32)/32).
  const int bb = row0 >> 11;                 // row0 / 2048
#pragma unroll
  for (int f = 0; f < 4; ++f) {
    const int n = n0 + f * 16 + m16;
    const int h = n >> 6;
    const int d = n & 63;
    float sc = 1.0f;
    if (z < 2) {
      const int jj = d & 31;
      float theta = exp2f(-(float)jj * 0.4152410118609203f); // log2(1e4)/32
      float ang = (float)bb * theta;
      float c = __cosf(ang), s = __sinf(ang);
      sc = (d < 32) ? (c + s) : (c - s);
    }
#pragma unroll
    for (int mi = 0; mi < 2; ++mi) {
      const int srow0 = (row0 & 2047) + 16 * mi + 8 * hi;
#pragma unroll
      for (int r = 0; r < 8; ++r) {
        const int srow = srow0 + r;
        if (z == 2) {
          Vt[(((size_t)bb * N_HEADS + h) * HEAD_DIM + d) * SEQ + srow] =
              (_Float16)acc[mi][f][r];
        } else {
          _Float16* dst = (z == 0) ? Qh : Kh;
          dst[(((size_t)bb * N_HEADS + h) * SEQ + srow) * HEAD_DIM + d] =
              (_Float16)(acc[mi][f][r] * sc);
        }
      }
    }
  }
}

// ---- Stage 3: flash attention --------------------------------------------
// grid: B*H*(SEQ/64) = 2048 blocks, block: 128 (4 waves).
// Each wave: 16 query rows, online softmax over 32-key blocks.
__global__ void attention(const _Float16* __restrict__ Qh,
                          const _Float16* __restrict__ Kh,
                          const _Float16* __restrict__ Vt,
                          float* __restrict__ out) {
  __shared__ __align__(64) _Float16 Plds[4][16][32];

  const int lane = threadIdx.x & 31;
  const int wave = threadIdx.x >> 5;
  const int m16 = lane & 15;
  const int hi = lane >> 4;

  const int bh = blockIdx.x >> 5;                       // b*16 + h
  const int q0 = ((blockIdx.x & 31) << 6) + wave * 16;  // query tile start

  const _Float16* Q = Qh + (size_t)bh * SEQ * HEAD_DIM;
  const _Float16* K = Kh + (size_t)bh * SEQ * HEAD_DIM;
  const _Float16* V = Vt + (size_t)bh * HEAD_DIM * SEQ;

  // Q fragments for K-dims d=0..31 and d=32..63 (held for whole loop)
  v16h aq0 = load_a_frag(Q, HEAD_DIM, q0 + m16, hi, 0);
  v16h aq1 = load_a_frag(Q, HEAD_DIM, q0 + m16, hi, 32);

  v8f o[4] = {};
  v8f mrow, lrow;
#pragma unroll
  for (int r = 0; r < 8; ++r) { mrow[r] = -__builtin_inff(); lrow[r] = 0.0f; }

  _Float16* pw = &Plds[wave][0][0];

  for (int j = 0; j < SEQ; j += 32) {
    // Prefetch next key block of K (32 rows) and next column block of Vt
    // (stays within the workspace; speculative prefetch is droppable).
    __builtin_prefetch(K + (size_t)(j + 32 + lane) * HEAD_DIM, 0, 0);
    __builtin_prefetch(V + (size_t)lane * SEQ + j + 32, 0, 0);
    __builtin_prefetch(V + (size_t)(32 + lane) * SEQ + j + 32, 0, 0);

    // --- scores: two 16x16 tiles (keys j..j+15 and j+16..j+31), K-dim = 64
    v8f s0 = {}, s1 = {};
    {
      v16h b;
      b = load_b_frag(K, HEAD_DIM, j + m16, hi, 0);       s0 = wmma_f16(aq0, b, s0);
      b = load_b_frag(K, HEAD_DIM, j + m16, hi, 32);      s0 = wmma_f16(aq1, b, s0);
      b = load_b_frag(K, HEAD_DIM, j + 16 + m16, hi, 0);  s1 = wmma_f16(aq0, b, s1);
      b = load_b_frag(K, HEAD_DIM, j + 16 + m16, hi, 32); s1 = wmma_f16(aq1, b, s1);
    }

    // --- online softmax stats. Each row lives in one 16-lane half per VGPR.
    v8f t, mnew, corr, p0, p1, rs;
#pragma unroll
    for (int r = 0; r < 8; ++r) {
      s0[r] *= 0.125f;  // 1/sqrt(64)
      s1[r] *= 0.125f;
      t[r] = fmaxf(s0[r], s1[r]);
    }
#pragma unroll
    for (int m = 1; m < 16; m <<= 1)
#pragma unroll
      for (int r = 0; r < 8; ++r) t[r] = fmaxf(t[r], __shfl_xor(t[r], m, 32));
#pragma unroll
    for (int r = 0; r < 8; ++r) {
      mnew[r] = fmaxf(mrow[r], t[r]);
      corr[r] = __expf(mrow[r] - mnew[r]);
      p0[r] = __expf(s0[r] - mnew[r]);
      p1[r] = __expf(s1[r] - mnew[r]);
      rs[r] = p0[r] + p1[r];
    }
#pragma unroll
    for (int m = 1; m < 16; m <<= 1)
#pragma unroll
      for (int r = 0; r < 8; ++r) rs[r] += __shfl_xor(rs[r], m, 32);
#pragma unroll
    for (int r = 0; r < 8; ++r) {
      lrow[r] = lrow[r] * corr[r] + rs[r];
      mrow[r] = mnew[r];
    }
#pragma unroll
    for (int f = 0; f < 4; ++f)
#pragma unroll
      for (int r = 0; r < 8; ++r) o[f][r] *= corr[r];

    // --- P (C-layout, lane<->col) -> LDS -> reload as A-fragment (lane<->row)
#pragma unroll
    for (int r = 0; r < 8; ++r) {
      pw[(r + 8 * hi) * 32 + m16]      = (_Float16)p0[r];
      pw[(r + 8 * hi) * 32 + 16 + m16] = (_Float16)p1[r];
    }
    asm volatile("s_wait_dscnt 0" ::: "memory");  // in-wave LDS RAW fence
    v16h ap = load_a_frag(pw, 32, m16, hi, 0);

    // --- O += P (16x32) x V (32x64): V stored transposed [d][s]
#pragma unroll
    for (int f = 0; f < 4; ++f) {
      v16h bv = load_b_frag(V, SEQ, f * 16 + m16, hi, j);
      o[f] = wmma_f16(ap, bv, o[f]);
    }
  }

  // --- normalize and write out [B,S,D] f32
  const int b = bh >> 4;
  const int h = bh & 15;
#pragma unroll
  for (int f = 0; f < 4; ++f) {
#pragma unroll
    for (int r = 0; r < 8; ++r) {
      const int qrow = q0 + r + 8 * hi;
      const int d = f * 16 + m16;
      out[((size_t)b * SEQ + qrow) * D_MODEL + h * HEAD_DIM + d] =
          o[f][r] / lrow[r];
    }
  }
}

// ---------------------------------------------------------------------------
extern "C" void kernel_launch(void* const* d_in, const int* in_sizes, int n_in,
                              void* d_out, int out_size, void* d_ws,
                              size_t ws_size, hipStream_t stream) {
  (void)in_sizes; (void)n_in; (void)out_size; (void)ws_size;
  const float* x  = (const float*)d_in[0];
  const float* Wq = (const float*)d_in[1];
  const float* Wk = (const float*)d_in[2];
  const float* Wv = (const float*)d_in[3];

  // workspace carve-up (f16): xh | Wh(q,k,v) | Qh | Kh | Vt   (~73.4 MB)
  _Float16* xh = (_Float16*)d_ws;
  _Float16* Wh = xh + (size_t)ROWS * D_MODEL;
  _Float16* Qh = Wh + (size_t)3 * D_MODEL * D_MODEL;
  _Float16* Kh = Qh + (size_t)ROWS * D_MODEL;
  _Float16* Vt = Kh + (size_t)ROWS * D_MODEL;

  const int nx4 = ROWS * D_MODEL / 4;
  const int nw4 = D_MODEL * D_MODEL / 4;
  cast_f32_f16<<<(nx4 + 255) / 256, 256, 0, stream>>>(x, xh, nx4);
  cast_f32_f16<<<(nw4 + 255) / 256, 256, 0, stream>>>(Wq, Wh, nw4);
  cast_f32_f16<<<(nw4 + 255) / 256, 256, 0, stream>>>(Wk, Wh + (size_t)D_MODEL * D_MODEL, nw4);
  cast_f32_f16<<<(nw4 + 255) / 256, 256, 0, stream>>>(Wv, Wh + (size_t)2 * D_MODEL * D_MODEL, nw4);

  dim3 ggrid(ROWS / 128, D_MODEL / 64, 3);
  qkv_gemm_rope<<<ggrid, 128, 0, stream>>>(xh, Wh, Qh, Kh, Vt);

  attention<<<BATCH * N_HEADS * (SEQ / 64), 128, 0, stream>>>(Qh, Kh, Vt,
                                                              (float*)d_out);
}
